// DeepseekV2MoE_52046413693388
// MI455X (gfx1250) — matrix operand order repaired and checked by reading, not compiled, not run on hardware
//
#include <hip/hip_runtime.h>
#include <hip/hip_bf16.h>

typedef __attribute__((ext_vector_type(16))) __bf16 v16bf;
typedef __attribute__((ext_vector_type(8)))  float  v8f;

#define T_TOK 1024
#define H_DIM 2048
#define E_EXP 64
#define I_DIM 1408
#define TOPK  6
#define SHI   2816
#define NASSIGN (T_TOK * TOPK)   // 6144

// ---- workspace layout (bytes) ----
#define OFF_COUNTS   0u
#define OFF_OFFSETS  256u
#define OFF_CURSOR   512u
#define OFF_TOPK_IDX 1024u                      // 1024*6*4 = 24576
#define OFF_TOPK_W   25600u                     // 24576
#define OFF_TOK_IDS  50176u                     // 6144*4 = 24576
#define OFF_TOK_W    74752u                     // 24576
#define OFF_HBUF     99328u                     // 6144*1408*2 = 17301504
#define OFF_ACT      17400832u                  // 1024*2816*2 = 5767168
// total ~23.2 MB

// ---------------- helpers ----------------

__device__ inline v16bf zero_v16bf() {
    v16bf z;
#pragma unroll
    for (int i = 0; i < 16; ++i) z[i] = (__bf16)0.0f;
    return z;
}

// A-matrix tile (16x32 bf16) for one wave, built from an fp32 row in memory.
// lane l (l%16 = row m): l<16 holds K = {k0..k0+7, k0+16..k0+23},
//                        l>=16 holds K = {k0+8..k0+15, k0+24..k0+31}.
__device__ inline v16bf load_a_f32(const float* __restrict__ row, int k0, int half) {
    v16bf a;
    const float* p0 = row + k0 + half * 8;
    const float* p1 = row + k0 + 16 + half * 8;
#pragma unroll
    for (int i = 0; i < 8; ++i) {
        a[i]     = (__bf16)p0[i];
        a[8 + i] = (__bf16)p1[i];
    }
    return a;
}

__device__ inline v16bf load_a_bf16(const __bf16* __restrict__ row, int k0, int half) {
    v16bf a;
    const __bf16* p0 = row + k0 + half * 8;
    const __bf16* p1 = row + k0 + 16 + half * 8;
#pragma unroll
    for (int i = 0; i < 8; ++i) {
        a[i]     = p0[i];
        a[8 + i] = p1[i];
    }
    return a;
}

// B-matrix tile (32x16 bf16): lane l holds column n = l%16,
// K = k0 + {8h..8h+7, 16+8h..16+8h+7}, h = l/16 (same order as A).
__device__ inline v16bf load_b_f32(const float* __restrict__ row, int k0, int half) {
    v16bf b;
    const float* p0 = row + k0 + half * 8;
    const float* p1 = row + k0 + 16 + half * 8;
#pragma unroll
    for (int i = 0; i < 8; ++i) {
        b[i]     = (__bf16)p0[i];
        b[8 + i] = (__bf16)p1[i];
    }
    return b;
}

// fast silu: v * rcp(1 + exp(-v))  -> v_exp_f32 + v_rcp_f32, no IEEE div chain
__device__ inline float silu_f(float v) {
    return v * __builtin_amdgcn_rcpf(1.0f + __expf(-v));
}

#define WMMA_BF16(A, B, C) \
    __builtin_amdgcn_wmma_f32_16x16x32_bf16(false, (A), false, (B), (short)0, (C), false, false)

// ---------------- kernel 1: reset counters ----------------
__global__ void k_reset(int* __restrict__ counts) {
    if (threadIdx.x < E_EXP) counts[threadIdx.x] = 0;
}

// ---------------- kernel 2: router (1 wave per token) ----------------
__global__ __launch_bounds__(32) void k_router(
    const float* __restrict__ x, const float* __restrict__ gate_w,
    int* __restrict__ counts, int* __restrict__ topk_idx, float* __restrict__ topk_w)
{
    const int t    = blockIdx.x;
    const int lane = threadIdx.x & 31;
    const float* xr = x + (size_t)t * H_DIM;
    const float* g0 = gate_w + (size_t)lane * H_DIM;
    const float* g1 = gate_w + (size_t)(lane + 32) * H_DIM;

    float acc0 = 0.f, acc1 = 0.f;
    for (int h = 0; h < H_DIM; ++h) {
        float xv = xr[h];
        acc0 = fmaf(xv, g0[h], acc0);
        acc1 = fmaf(xv, g1[h], acc1);
    }
    // softmax over 64 logits (2 per lane)
    float m = fmaxf(acc0, acc1);
#pragma unroll
    for (int o = 16; o > 0; o >>= 1) m = fmaxf(m, __shfl_xor(m, o, 32));
    float e0 = __expf(acc0 - m), e1 = __expf(acc1 - m);
    float s = e0 + e1;
#pragma unroll
    for (int o = 16; o > 0; o >>= 1) s += __shfl_xor(s, o, 32);
    float rs = __builtin_amdgcn_rcpf(s);
    float p0 = e0 * rs, p1 = e1 * rs;

    // top-6 via iterated wave argmax
    float w[TOPK]; int id[TOPK];
#pragma unroll
    for (int k = 0; k < TOPK; ++k) {
        float best = p0; int bi = lane;
        if (p1 > best) { best = p1; bi = lane + 32; }
#pragma unroll
        for (int o = 16; o > 0; o >>= 1) {
            float ob = __shfl_xor(best, o, 32);
            int  obi = __shfl_xor(bi, o, 32);
            if (ob > best || (ob == best && obi < bi)) { best = ob; bi = obi; }
        }
        w[k] = best; id[k] = bi;
        if (bi == lane)      p0 = -1.0f;
        if (bi == lane + 32) p1 = -1.0f;
    }
    float sw = 0.f;
#pragma unroll
    for (int k = 0; k < TOPK; ++k) sw += w[k];
    float rsw = __builtin_amdgcn_rcpf(sw);
    if (lane == 0) {
#pragma unroll
        for (int k = 0; k < TOPK; ++k) {
            topk_idx[t * TOPK + k] = id[k];
            topk_w[t * TOPK + k]   = w[k] * rsw;
            atomicAdd(&counts[id[k]], 1);
        }
    }
}

// ---------------- kernel 3: 64-entry exclusive scan ----------------
__global__ void k_scan(const int* __restrict__ counts, int* __restrict__ offsets,
                       int* __restrict__ cursor)
{
    if (threadIdx.x == 0) {
        int acc = 0;
        for (int e = 0; e < E_EXP; ++e) {
            offsets[e] = acc;
            cursor[e]  = acc;
            acc += counts[e];
        }
    }
}

// ---------------- kernel 4: scatter tokens into per-expert lists ----------------
__global__ void k_scatter(const int* __restrict__ topk_idx, const float* __restrict__ topk_w,
                          int* __restrict__ cursor, int* __restrict__ tok_ids,
                          float* __restrict__ tok_w)
{
    int t = blockIdx.x * blockDim.x + threadIdx.x;
    if (t >= T_TOK) return;
#pragma unroll
    for (int k = 0; k < TOPK; ++k) {
        int e   = topk_idx[t * TOPK + k];
        float w = topk_w[t * TOPK + k];
        int pos = atomicAdd(&cursor[e], 1);
        tok_ids[pos] = t;
        tok_w[pos]   = w;
    }
}

// ---------------- kernel 5: shared expert gate_up + SwiGLU ----------------
// grid (T/16, SHI/128), 128 thr (4 waves); each wave: two 16x16 g-tiles + two u-tiles
// sharing one A fragment. Four distinct B fragments per K-step so post-WMMA cvts
// are independent VALU (fills the 4 co-exec hazard slots, no v_nops).
__global__ __launch_bounds__(128) void k_shared_gu(
    const float* __restrict__ x, const float* __restrict__ sgu,
    __bf16* __restrict__ act_buf)
{
    const int lane = threadIdx.x & 31;
    const int wv   = threadIdx.x >> 5;
    const int half = lane >> 4;
    const int ln   = lane & 15;
    const int mrow = blockIdx.x * 16 + ln;          // token row this lane feeds as A
    const int nc0  = blockIdx.y * 128 + wv * 32 + ln;
    const int nc1  = nc0 + 16;

    const float* arow  = x   + (size_t)mrow * H_DIM;
    const float* grow0 = sgu + (size_t)nc0 * H_DIM;
    const float* grow1 = sgu + (size_t)nc1 * H_DIM;
    const float* urow0 = sgu + (size_t)(SHI + nc0) * H_DIM;
    const float* urow1 = sgu + (size_t)(SHI + nc1) * H_DIM;

    v8f aG0 = {}, aU0 = {}, aG1 = {}, aU1 = {};
    for (int k0 = 0; k0 < H_DIM; k0 += 32) {
        __builtin_prefetch(grow0 + k0 + 64, 0, 0);
        __builtin_prefetch(urow0 + k0 + 64, 0, 0);
        v16bf a   = load_a_f32(arow, k0, half);
        v16bf bg0 = load_b_f32(grow0, k0, half);
        v16bf bg1 = load_b_f32(grow1, k0, half);
        v16bf bu0 = load_b_f32(urow0, k0, half);
        v16bf bu1 = load_b_f32(urow1, k0, half);
        aG0 = WMMA_BF16(a, bg0, aG0);
        aG1 = WMMA_BF16(a, bg1, aG1);
        aU0 = WMMA_BF16(a, bu0, aU0);
        aU1 = WMMA_BF16(a, bu1, aU1);
    }
#pragma unroll
    for (int r = 0; r < 8; ++r) {
        int m = blockIdx.x * 16 + r + half * 8;     // C layout: M = r + 8*(lane/16)
        act_buf[(size_t)m * SHI + nc0] = (__bf16)(silu_f(aG0[r]) * aU0[r]);
        act_buf[(size_t)m * SHI + nc1] = (__bf16)(silu_f(aG1[r]) * aU1[r]);
    }
}

// ---------------- kernel 6: shared expert down-proj (plain store -> d_out) ----------------
// grid (T/16, H/128), 128 thr (4 waves), 2 N-subtiles per wave, K unrolled x2
__global__ __launch_bounds__(128) void k_shared_down(
    const __bf16* __restrict__ act_buf, const float* __restrict__ sdown,
    float* __restrict__ out)
{
    const int lane = threadIdx.x & 31;
    const int wv   = threadIdx.x >> 5;
    const int half = lane >> 4;
    const int ln   = lane & 15;
    const int mrow = blockIdx.x * 16 + ln;
    const int nc0  = blockIdx.y * 128 + wv * 32 + ln;
    const int nc1  = nc0 + 16;

    const __bf16* arow  = act_buf + (size_t)mrow * SHI;
    const float*  brow0 = sdown   + (size_t)nc0 * SHI;
    const float*  brow1 = sdown   + (size_t)nc1 * SHI;

    v8f acc0 = {}, acc1 = {};
    for (int k0 = 0; k0 < SHI; k0 += 64) {          // SHI = 44 * 64
        __builtin_prefetch(brow0 + k0 + 128, 0, 0);
        __builtin_prefetch(brow1 + k0 + 128, 0, 0);
        v16bf aA  = load_a_bf16(arow, k0, half);
        v16bf bA0 = load_b_f32(brow0, k0, half);
        v16bf bA1 = load_b_f32(brow1, k0, half);
        acc0 = WMMA_BF16(aA, bA0, acc0);
        acc1 = WMMA_BF16(aA, bA1, acc1);
        v16bf aB  = load_a_bf16(arow, k0 + 32, half);
        v16bf bB0 = load_b_f32(brow0, k0 + 32, half);
        v16bf bB1 = load_b_f32(brow1, k0 + 32, half);
        acc0 = WMMA_BF16(aB, bB0, acc0);
        acc1 = WMMA_BF16(aB, bB1, acc1);
    }
#pragma unroll
    for (int r = 0; r < 8; ++r) {
        int m = blockIdx.x * 16 + r + half * 8;
        out[(size_t)m * H_DIM + nc0] = acc0[r];
        out[(size_t)m * H_DIM + nc1] = acc1[r];
    }
}

// ---------------- kernel 7: routed experts up-proj + SiLU ----------------
// grid (E, I/128), 128 thr (4 waves), 2 N-subtiles per wave, K unrolled x2;
// loops over this expert's 16-token M-tiles (weight slice re-reads hit L2).
__global__ __launch_bounds__(128) void k_moe_up(
    const float* __restrict__ x, const float* __restrict__ w1,
    const int* __restrict__ counts, const int* __restrict__ offsets,
    const int* __restrict__ tok_ids, __bf16* __restrict__ h_buf)
{
    const int e   = blockIdx.x;
    const int cnt = counts[e];
    if (cnt == 0) return;
    const int off  = offsets[e];
    const int lane = threadIdx.x & 31;
    const int wv   = threadIdx.x >> 5;
    const int half = lane >> 4;
    const int ln   = lane & 15;
    const int nc0  = blockIdx.y * 128 + wv * 32 + ln;
    const int nc1  = nc0 + 16;
    const float* brow0 = w1 + ((size_t)e * I_DIM + nc0) * H_DIM;
    const float* brow1 = w1 + ((size_t)e * I_DIM + nc1) * H_DIM;

    const int mtiles = (cnt + 15) >> 4;
    for (int mt = 0; mt < mtiles; ++mt) {
        const int mrow = mt * 16 + ln;
        const bool valid = (mrow < cnt);
        const int tok = valid ? tok_ids[off + mrow] : 0;
        const float* arow = x + (size_t)tok * H_DIM;

        v8f acc0 = {}, acc1 = {};
        for (int k0 = 0; k0 < H_DIM; k0 += 64) {    // H = 32 * 64
            __builtin_prefetch(brow0 + k0 + 128, 0, 0);
            __builtin_prefetch(brow1 + k0 + 128, 0, 0);
            v16bf aA  = valid ? load_a_f32(arow, k0, half) : zero_v16bf();
            v16bf bA0 = load_b_f32(brow0, k0, half);
            v16bf bA1 = load_b_f32(brow1, k0, half);
            acc0 = WMMA_BF16(aA, bA0, acc0);
            acc1 = WMMA_BF16(aA, bA1, acc1);
            v16bf aB  = valid ? load_a_f32(arow, k0 + 32, half) : zero_v16bf();
            v16bf bB0 = load_b_f32(brow0, k0 + 32, half);
            v16bf bB1 = load_b_f32(brow1, k0 + 32, half);
            acc0 = WMMA_BF16(aB, bB0, acc0);
            acc1 = WMMA_BF16(aB, bB1, acc1);
        }
#pragma unroll
        for (int r = 0; r < 8; ++r) {
            int gm = mt * 16 + r + half * 8;
            if (gm < cnt) {
                h_buf[(size_t)(off + gm) * I_DIM + nc0] = (__bf16)silu_f(acc0[r]);
                h_buf[(size_t)(off + gm) * I_DIM + nc1] = (__bf16)silu_f(acc1[r]);
            }
        }
    }
}

// ---------------- kernel 8: routed experts down-proj + weighted atomic combine ----------------
// grid (E, H/128), 128 thr (4 waves), 2 N-subtiles per wave, K unrolled x2
__global__ __launch_bounds__(128) void k_moe_down(
    const __bf16* __restrict__ h_buf, const float* __restrict__ w2,
    const int* __restrict__ counts, const int* __restrict__ offsets,
    const int* __restrict__ tok_ids, const float* __restrict__ tok_w,
    float* __restrict__ out)
{
    const int e   = blockIdx.x;
    const int cnt = counts[e];
    if (cnt == 0) return;
    const int off  = offsets[e];
    const int lane = threadIdx.x & 31;
    const int wv   = threadIdx.x >> 5;
    const int half = lane >> 4;
    const int ln   = lane & 15;
    const int nc0  = blockIdx.y * 128 + wv * 32 + ln;
    const int nc1  = nc0 + 16;
    const float* brow0 = w2 + ((size_t)e * H_DIM + nc0) * I_DIM;
    const float* brow1 = w2 + ((size_t)e * H_DIM + nc1) * I_DIM;

    const int mtiles = (cnt + 15) >> 4;
    for (int mt = 0; mt < mtiles; ++mt) {
        const int mrow = mt * 16 + ln;
        const bool valid = (mrow < cnt);
        const __bf16* arow = h_buf + (size_t)(off + (valid ? mrow : 0)) * I_DIM;

        v8f acc0 = {}, acc1 = {};
        for (int k0 = 0; k0 < I_DIM; k0 += 64) {    // I = 22 * 64
            __builtin_prefetch(brow0 + k0 + 128, 0, 0);
            __builtin_prefetch(brow1 + k0 + 128, 0, 0);
            v16bf aA  = valid ? load_a_bf16(arow, k0, half) : zero_v16bf();
            v16bf bA0 = load_b_f32(brow0, k0, half);
            v16bf bA1 = load_b_f32(brow1, k0, half);
            acc0 = WMMA_BF16(aA, bA0, acc0);
            acc1 = WMMA_BF16(aA, bA1, acc1);
            v16bf aB  = valid ? load_a_bf16(arow, k0 + 32, half) : zero_v16bf();
            v16bf bB0 = load_b_f32(brow0, k0 + 32, half);
            v16bf bB1 = load_b_f32(brow1, k0 + 32, half);
            acc0 = WMMA_BF16(aB, bB0, acc0);
            acc1 = WMMA_BF16(aB, bB1, acc1);
        }
#pragma unroll
        for (int r = 0; r < 8; ++r) {
            int gm = mt * 16 + r + half * 8;
            if (gm < cnt) {
                int   tok = tok_ids[off + gm];
                float w   = tok_w[off + gm];
                atomicAdd(&out[(size_t)tok * H_DIM + nc0], acc0[r] * w);
                atomicAdd(&out[(size_t)tok * H_DIM + nc1], acc1[r] * w);
            }
        }
    }
}

// ---------------- launcher ----------------
extern "C" void kernel_launch(void* const* d_in, const int* in_sizes, int n_in,
                              void* d_out, int out_size, void* d_ws, size_t ws_size,
                              hipStream_t stream) {
    const float* x     = (const float*)d_in[0];   // [T, H]
    const float* gatew = (const float*)d_in[1];   // [E, H]
    const float* w1    = (const float*)d_in[2];   // [E, I, H]
    const float* w2    = (const float*)d_in[3];   // [E, H, I]
    const float* sgu   = (const float*)d_in[4];   // [2*SHI, H]
    const float* sdown = (const float*)d_in[5];   // [H, SHI]
    float* out         = (float*)d_out;           // [T, H]

    char* ws = (char*)d_ws;
    int*    counts   = (int*)  (ws + OFF_COUNTS);
    int*    offsets  = (int*)  (ws + OFF_OFFSETS);
    int*    cursor   = (int*)  (ws + OFF_CURSOR);
    int*    topk_idx = (int*)  (ws + OFF_TOPK_IDX);
    float*  topk_w   = (float*)(ws + OFF_TOPK_W);
    int*    tok_ids  = (int*)  (ws + OFF_TOK_IDS);
    float*  tok_w    = (float*)(ws + OFF_TOK_W);
    __bf16* h_buf    = (__bf16*)(ws + OFF_HBUF);
    __bf16* act_buf  = (__bf16*)(ws + OFF_ACT);

    k_reset<<<1, 64, 0, stream>>>(counts);
    k_router<<<T_TOK, 32, 0, stream>>>(x, gatew, counts, topk_idx, topk_w);
    k_scan<<<1, 32, 0, stream>>>(counts, offsets, cursor);
    k_scatter<<<(T_TOK + 255) / 256, 256, 0, stream>>>(topk_idx, topk_w, cursor, tok_ids, tok_w);

    // shared expert: establishes d_out with plain stores
    k_shared_gu<<<dim3(T_TOK / 16, SHI / 128), 128, 0, stream>>>(x, sgu, act_buf);
    k_shared_down<<<dim3(T_TOK / 16, H_DIM / 128), 128, 0, stream>>>(act_buf, sdown, out);

    // routed experts: atomic-add on top
    k_moe_up<<<dim3(E_EXP, I_DIM / 128), 128, 0, stream>>>(x, w1, counts, offsets, tok_ids, h_buf);
    k_moe_down<<<dim3(E_EXP, H_DIM / 128), 128, 0, stream>>>(h_buf, w2, counts, offsets, tok_ids, tok_w, out);
}
